// MultiTimescaleLIFBlock_27616639713852
// MI455X (gfx1250) — matrix operand order repaired and checked by reading, not compiled, not run on hardware
//
#include <hip/hip_runtime.h>

typedef __attribute__((ext_vector_type(16))) _Float16 v16h;
typedef __attribute__((ext_vector_type(8)))  _Float16 v8h;
typedef __attribute__((ext_vector_type(8)))  float    v8f;

#define T_STEPS 1024
#define BATCH   128
#define IN_DIM  256
#define HID     512
#define OUT_DIM 256
#define K_FUSE  (3*HID)           // 1536

// ---------------- workspace layout (bytes) ----------------
#define XPROJ_OFF     0ULL
#define XPROJ_BYTES   ((size_t)T_STEPS*BATCH*HID*4)       // 268,435,456
#define WPROJ16_OFF   (XPROJ_OFF + XPROJ_BYTES)
#define WPROJ16_BYTES ((size_t)IN_DIM*HID*2)
#define WFUSE16_OFF   (WPROJ16_OFF + WPROJ16_BYTES)
#define WFUSE16_BYTES ((size_t)K_FUSE*HID*2)
#define WHEAD16_OFF   (WFUSE16_OFF + WFUSE16_BYTES)
#define WHEAD16_BYTES ((size_t)HID*OUT_DIM*2)
#define X16_OFF       (WHEAD16_OFF + WHEAD16_BYTES)
#define X16_BYTES     ((size_t)T_STEPS*BATCH*IN_DIM*2)    // 67,108,864

// ---------------- output layout (floats) ----------------
#define READOUT_ELEMS ((size_t)BATCH*OUT_DIM)                 // 32768
#define LOGITS_ELEMS  ((size_t)T_STEPS*BATCH*OUT_DIM)         // 33,554,432
#define SR_OFF_ELEMS  (READOUT_ELEMS + LOGITS_ELEMS)

// ---- WMMA fragment helpers (contiguous k = 16h + s, wave32) ----
// A 16x32 f16 from LDS: each lane reads 16 contiguous k (k = 16h + s) as two 16B runs (ds_load_b128 x2).
__device__ __forceinline__ v16h frag_a(const _Float16* row, int kb_a) {
    v8h lo = *(const v8h*)(row + kb_a);
    v8h hi = *(const v8h*)(row + kb_a + 8);
    return __builtin_shufflevector(lo, hi, 0,1,2,3,4,5,6,7,8,9,10,11,12,13,14,15);
}
// B 32x16 f16 from fragment-packed global memory: 32 contiguous bytes per lane.
__device__ __forceinline__ v16h frag_b_pk(const _Float16* blockbase, int lane) {
    return *(const v16h*)(blockbase + lane * 16);
}

// pack one element of a row-major [K][N] f32 matrix into fragment-packed f16:
// block (kt,nt) of 32Kx16N holds 512 halfs; lane l = ((kin>=16)<<4)|nin owns
// 16 contiguous halfs, element e = kin&15.  (matches frag_b_pk / frag_a's contiguous k = 16h + e order)
__device__ __forceinline__ void pack_one(const float* __restrict__ src,
                                         _Float16* __restrict__ dst,
                                         int N, int i) {
    const int k = i / N, n = i - k * N;
    const int kt = k >> 5, kin = k & 31;
    const int nt = n >> 4, nin = n & 15;
    const int lane = ((kin >> 4) << 4) | nin;
    const int e = kin & 15;
    dst[((size_t)(kt * (N >> 4) + nt)) * 512 + lane * 16 + e] = (_Float16)src[i];
}

// ---------------- kernel 0a: pack weights f32 -> fragment-packed f16 ----------------
__global__ void prep_weights(const float* __restrict__ Wp, const float* __restrict__ Wf,
                             const float* __restrict__ Wh,
                             _Float16* wp16, _Float16* wf16, _Float16* wh16) {
    const int n1 = IN_DIM*HID, n2 = K_FUSE*HID, n3 = HID*OUT_DIM;
    for (int i = blockIdx.x*blockDim.x + threadIdx.x; i < n2; i += gridDim.x*blockDim.x) {
        if (i < n1) pack_one(Wp, wp16, HID, i);
        pack_one(Wf, wf16, HID, i);
        if (i < n3) pack_one(Wh, wh16, OUT_DIM, i);
    }
}

// ---------------- kernel 0b: x_seq f32 -> f16 (feeds async-to-LDS staging) --------
__global__ __launch_bounds__(256) void conv_x(const float* __restrict__ x,
                                              _Float16* __restrict__ x16) {
    const size_t i = ((size_t)blockIdx.x*blockDim.x + threadIdx.x) * 8;
    const float4 a = *(const float4*)(x + i);
    const float4 b = *(const float4*)(x + i + 4);
    v8h o;
    o[0]=(_Float16)a.x; o[1]=(_Float16)a.y; o[2]=(_Float16)a.z; o[3]=(_Float16)a.w;
    o[4]=(_Float16)b.x; o[5]=(_Float16)b.y; o[6]=(_Float16)b.z; o[7]=(_Float16)b.w;
    *(v8h*)(x16 + i) = o;
}

__global__ void init_sr(float* sr) {
    if (threadIdx.x < 3) sr[threadIdx.x] = 0.0f;
}

// ---------------- kernel 1: Xproj = x_seq @ W_proj + b_proj ----------------
// Grid: 8192 WGs, one 16-row M tile. A tile staged via global_load_async_to_lds_b128.
#define ATILE_LD 264   // halfs; 528B/row = 132 dwords = 4 mod 64 banks -> conflict-free

__global__ __launch_bounds__(256) void proj_kernel(const _Float16* __restrict__ x16,
                                                   const _Float16* __restrict__ w16p,
                                                   const float* __restrict__ bias,
                                                   float* __restrict__ xproj) {
    __shared__ __align__(16) _Float16 atile[16 * ATILE_LD];
    const size_t row0 = (size_t)blockIdx.x * 16;
    const int tid = threadIdx.x;

    // ---- async stage 16x256 f16 tile (8KB) into LDS, 16B per lane-op ----
    {
        const char* gsrc = (const char*)(x16 + row0 * IN_DIM);
        const unsigned ldsbase = (unsigned)(size_t)(&atile[0]);
#pragma unroll
        for (int it = 0; it < 2; ++it) {
            const int xfer = tid + it * 256;            // 512 x 16B transfers
            const int r = xfer >> 5, c = xfer & 31;     // 32 per row
            const unsigned loff = ldsbase + (unsigned)(r * (ATILE_LD * 2) + c * 16);
            const char* gaddr = gsrc + (size_t)r * (IN_DIM * 2) + c * 16;
            asm volatile("global_load_async_to_lds_b128 %0, %1, off"
                         :: "v"(loff), "v"(gaddr) : "memory");
        }
        asm volatile("s_wait_asynccnt 0x0" ::: "memory");
    }
    __syncthreads();

    const int wave  = tid >> 5;
    const int lane  = tid & 31;
    const int mrow  = lane & 15;
    const int kb_a  = (lane >> 4) << 4;
    const int nlane = lane & 15;
    const int mhalf = (lane >> 4) << 3;

#pragma unroll
    for (int j = 0; j < 4; ++j) {
        const int nt = wave*4 + j;                 // N tile 0..31
        const int n0 = nt * 16;
        v8f acc = {};
        for (int kk = 0; kk < IN_DIM; kk += 32) {
            v16h a = frag_a(&atile[mrow*ATILE_LD + kk], kb_a);
            v16h b = frag_b_pk(w16p + (size_t)((kk >> 5)*(HID >> 4) + nt)*512, lane);
            acc = __builtin_amdgcn_wmma_f32_16x16x32_f16(false, a, false, b,
                                                         (short)0, acc, false, false);
        }
        const float bv = bias[n0 + nlane];
#pragma unroll
        for (int r = 0; r < 8; ++r)
            xproj[(row0 + mhalf + r)*HID + n0 + nlane] = acc[r] + bv;
    }
}

// ---------------- kernel 2: sequential LIF scan + fuse/head GEMMs ----------------
// Grid: 8 WGs (one per 16-batch-row tile) x 1024 threads (32 waves).
#define SPK_LD 1544   // halfs; 3088B/row = 772 dwords = 4 mod 64 banks
#define FUS_LD 520    // halfs; 1040B/row = 260 dwords = 4 mod 64 banks

__global__ __launch_bounds__(1024) void scan_kernel(const float* __restrict__ xproj,
                                                    const _Float16* __restrict__ wf16p,
                                                    const float* __restrict__ b_fuse,
                                                    const _Float16* __restrict__ wh16p,
                                                    const float* __restrict__ b_head,
                                                    float* __restrict__ out) {
    __shared__ __align__(16) _Float16 smem[16 * SPK_LD]; // spikes; overlaid by fused
    __shared__ float sr_red[3];

    const int b0   = blockIdx.x * 16;
    const int tid  = threadIdx.x;
    const int wave = tid >> 5;
    const int lane = tid & 31;

    const int mrow  = lane & 15;
    const int kb_a  = (lane >> 4) << 4;
    const int nlane = lane & 15;
    const int mhalf = (lane >> 4) << 3;

    const int   n0f   = wave * 16;               // fuse N tile (32 waves x 16 = 512)
    const float biasf = b_fuse[n0f + nlane];
    const int   n0h   = (wave & 15) * 16;        // head N tile (waves 0..15)
    const float biash = b_head[n0h + nlane];

    // each thread owns 8 CONTIGUOUS (b,h) elements: idx = tid*8 + j
    const int myb  = tid >> 6;                   // (tid*8)>>9
    const int myh0 = (tid & 63) * 8;

    float vf[8] = {}, vm[8] = {}, vs[8] = {};
    float cf = 0.f, cm = 0.f, cs = 0.f;
    v8f racc = {};

    float* readout = out;
    float* logits  = out + READOUT_ELEMS;

    for (int t = 0; t < T_STEPS; ++t) {
        // ---- LIF recurrence: spike from current V, then soft-reset integrate ----
        const float* xp = xproj + ((size_t)t*BATCH + b0)*HID + tid*8;
        const float4 xa = *(const float4*)(xp);
        const float4 xb = *(const float4*)(xp + 4);
        const float xv[8] = { xa.x, xa.y, xa.z, xa.w, xb.x, xb.y, xb.z, xb.w };
        v8h sfv, smv, ssv;
#pragma unroll
        for (int j = 0; j < 8; ++j) {
            const float sf = (vf[j] > 1.0f) ? 1.0f : 0.0f;
            const float sm = (vm[j] > 1.0f) ? 1.0f : 0.0f;
            const float ss = (vs[j] > 1.0f) ? 1.0f : 0.0f;
            vf[j] = 0.60f*vf[j] + xv[j] - sf;
            vm[j] = 0.80f*vm[j] + xv[j] - sm;
            vs[j] = 0.95f*vs[j] + xv[j] - ss;
            cf += sf; cm += sm; cs += ss;
            sfv[j] = (_Float16)sf; smv[j] = (_Float16)sm; ssv[j] = (_Float16)ss;
        }
        _Float16* srow = &smem[myb*SPK_LD + myh0];
        *(v8h*)(srow)           = sfv;           // ds_store_b128 x3
        *(v8h*)(srow + HID)     = smv;
        *(v8h*)(srow + 2*HID)   = ssv;
        __syncthreads();

        // ---- fuse GEMM: [16,1536] spikes @ [1536,512] W_fuse (packed) ----
        v8f facc = {};
        for (int kk = 0; kk < K_FUSE; kk += 32) {
            const size_t blk = (size_t)((kk >> 5)*(HID >> 4) + wave) * 512;
            if (kk + 32 < K_FUSE)
                __builtin_prefetch(wf16p + blk + (HID >> 4)*512 + lane*16, 0, 3);
            v16h a = frag_a(&smem[mrow*SPK_LD + kk], kb_a);
            v16h b = frag_b_pk(wf16p + blk, lane);
            facc = __builtin_amdgcn_wmma_f32_16x16x32_f16(false, a, false, b,
                                                          (short)0, facc, false, false);
        }
        __syncthreads();   // all waves done reading spikes; overlay fused activations
#pragma unroll
        for (int r = 0; r < 8; ++r)
            smem[(mhalf + r)*FUS_LD + n0f + nlane] = (_Float16)(facc[r] + biasf);
        __syncthreads();

        // ---- head GEMM: [16,512] fused @ [512,256] W_head (waves 0..15) ----
        if (wave < 16) {
            v8f hacc = {};
            for (int kk = 0; kk < HID; kk += 32) {
                v16h a = frag_a(&smem[mrow*FUS_LD + kk], kb_a);
                v16h b = frag_b_pk(wh16p +
                          (size_t)((kk >> 5)*(OUT_DIM >> 4) + (wave & 15))*512, lane);
                hacc = __builtin_amdgcn_wmma_f32_16x16x32_f16(false, a, false, b,
                                                              (short)0, hacc, false, false);
            }
            float* lrow = logits + ((size_t)t*BATCH + b0)*OUT_DIM;
#pragma unroll
            for (int r = 0; r < 8; ++r) {
                const float val = hacc[r] + biash;
                lrow[(size_t)(mhalf + r)*OUT_DIM + n0h + nlane] = val;
                racc[r] += val;
            }
        }
        __syncthreads();   // protect smem before next step's spike writes
    }

    // ---- readout = mean over T (each WG exclusively owns its batch rows) ----
    if (wave < 16) {
#pragma unroll
        for (int r = 0; r < 8; ++r)
            readout[(size_t)(b0 + mhalf + r)*OUT_DIM + n0h + nlane] =
                racc[r] * (1.0f / T_STEPS);
    }

    // ---- spike-rate means: block reduce then global atomics ----
    if (tid == 0) { sr_red[0] = 0.f; sr_red[1] = 0.f; sr_red[2] = 0.f; }
    __syncthreads();
    atomicAdd(&sr_red[0], cf);
    atomicAdd(&sr_red[1], cm);
    atomicAdd(&sr_red[2], cs);
    __syncthreads();
    if (tid == 0) {
        const float sc = 1.0f / ((float)T_STEPS * (float)BATCH * (float)HID);
        float* sr = out + SR_OFF_ELEMS;
        atomicAdd(&sr[0], sr_red[0] * sc);
        atomicAdd(&sr[1], sr_red[1] * sc);
        atomicAdd(&sr[2], sr_red[2] * sc);
    }
}

extern "C" void kernel_launch(void* const* d_in, const int* in_sizes, int n_in,
                              void* d_out, int out_size, void* d_ws, size_t ws_size,
                              hipStream_t stream) {
    const float* x_seq  = (const float*)d_in[0];
    const float* W_proj = (const float*)d_in[1];
    const float* b_proj = (const float*)d_in[2];
    const float* W_fuse = (const float*)d_in[3];
    const float* b_fuse = (const float*)d_in[4];
    const float* W_head = (const float*)d_in[5];
    const float* b_head = (const float*)d_in[6];

    char* ws = (char*)d_ws;
    float*    xproj = (float*)   (ws + XPROJ_OFF);
    _Float16* wp16  = (_Float16*)(ws + WPROJ16_OFF);
    _Float16* wf16  = (_Float16*)(ws + WFUSE16_OFF);
    _Float16* wh16  = (_Float16*)(ws + WHEAD16_OFF);
    _Float16* x16   = (_Float16*)(ws + X16_OFF);

    float* outf = (float*)d_out;

    conv_x<<<(T_STEPS*BATCH*IN_DIM)/(256*8), 256, 0, stream>>>(x_seq, x16);
    prep_weights<<<3072, 256, 0, stream>>>(W_proj, W_fuse, W_head, wp16, wf16, wh16);
    init_sr<<<1, 32, 0, stream>>>(outf + SR_OFF_ELEMS);
    proj_kernel<<<(T_STEPS*BATCH)/16, 256, 0, stream>>>(x16, wp16, b_proj, xproj);
    scan_kernel<<<BATCH/16, 1024, 0, stream>>>(xproj, wf16, b_fuse, wh16, b_head, outf);
}
